// StyleDiscriminator_73967926772458
// MI455X (gfx1250) — matrix operand order repaired and checked by reading, not compiled, not run on hardware
//
#include <hip/hip_runtime.h>
#include <hip/hip_bf16.h>
#include <math.h>
#include <stdint.h>

#define BB   256
#define TT   256
#define EMBD 512
#define HH   512
#define G4   2048   // 4*H
#define STY  128
#define D1N  1024
#define D2N  512
#define FK   1152   // 2H + STY
#define KC   64     // K-chunk staged in LDS per double-buffer slot

typedef __attribute__((ext_vector_type(16))) __bf16 v16bf;
typedef __attribute__((ext_vector_type(8)))  float  v8f;

struct Q32 { uint4 a, b; };

__device__ __forceinline__ unsigned short f2bf(float f) {
    unsigned u = __float_as_uint(f);
    u += 0x7FFFu + ((u >> 16) & 1u);
    return (unsigned short)(u >> 16);
}
__device__ __forceinline__ float bf2f(unsigned short h) {
    return __uint_as_float(((unsigned)h) << 16);
}
__device__ __forceinline__ float sigf(float x) { return 1.0f / (1.0f + __expf(-x)); }

// A-fragment: 16x32 bf16, row-major source, row stride ld (elements).
// lane l: M = l&15 ; K = kb..kb+15 (contiguous), kb = (l>=16)*16
__device__ __forceinline__ v16bf load_a_frag(const unsigned short* __restrict__ row0,
                                             int ld, int lane) {
    int m  = lane & 15;
    int kb = (lane >> 4) << 4;
    const unsigned short* p = row0 + (size_t)m * ld + kb;
    Q32 q;
    q.a = *reinterpret_cast<const uint4*>(p);
    q.b = *reinterpret_cast<const uint4*>(p + 8);
    return __builtin_bit_cast(v16bf, q);
}

// B-fragment: 32x16 bf16 from W^T stored [N,K] row-major (ldk = K).
// lane l: N = l&15 ; K = k0 + (l>=16)*16 .. +15 (contiguous)
__device__ __forceinline__ v16bf load_b_frag(const unsigned short* __restrict__ bt,
                                             int ldk, int n0, int k0, int lane) {
    int n  = lane & 15;
    int kb = (lane >> 4) << 4;
    const uint4* p = reinterpret_cast<const uint4*>(bt + (size_t)(n0 + n) * ldk + k0 + kb);
    Q32 q; q.a = p[0]; q.b = p[1];
    return __builtin_bit_cast(v16bf, q);
}

__device__ __forceinline__ v8f wmma_bf16(v16bf a, v16bf b, v8f c) {
    return __builtin_amdgcn_wmma_f32_16x16x32_bf16(false, a, false, b, (short)0, c, false, false);
}

// CDNA5 async copy: global -> LDS, 16B per lane, tracked by ASYNCcnt.
__device__ __forceinline__ void async_ld_b128(unsigned lds_off, const unsigned short* gptr) {
    asm volatile("global_load_async_to_lds_b128 %0, %1, off"
                 :: "v"(lds_off), "v"(gptr) : "memory");
}
__device__ __forceinline__ void wait_asynccnt0() {
    asm volatile("s_wait_asynccnt 0" ::: "memory");
}

// ---------------- weight cast+transpose: in [K,N] f32 -> out [N,K] bf16 ----------
__global__ __launch_bounds__(256) void castT_kernel(const float* __restrict__ in,
                                                    unsigned short* __restrict__ out,
                                                    int K, int N) {
    size_t i = (size_t)blockIdx.x * blockDim.x + threadIdx.x;
    if (i >= (size_t)K * N) return;
    int n = (int)(i / K);
    int k = (int)(i % K);
    out[i] = f2bf(in[(size_t)k * N + n]);
}

// ---------------- embedding gather -> X1 [T,B,EMB] bf16, mask [T,B] u8 ----------
__global__ __launch_bounds__(256) void embed_kernel(const int* __restrict__ seq,
                                                    const float* __restrict__ tok_emb,
                                                    unsigned short* __restrict__ X1,
                                                    unsigned char* __restrict__ maskT) {
    size_t i = (size_t)blockIdx.x * blockDim.x + threadIdx.x;
    if (i >= (size_t)BB * TT * EMBD) return;
    int e  = (int)(i & (EMBD - 1));
    size_t tb = i >> 9;                  // EMBD = 512
    int b = (int)(tb & (BB - 1));
    int t = (int)(tb >> 8);              // BB = 256
    int tokid = seq[(size_t)b * TT + t];
    X1[i] = f2bf(tok_emb[(size_t)tokid * EMBD + e]);
    if (e == 0) maskT[tb] = (tokid != 0) ? 1 : 0;
}

__global__ __launch_bounds__(256) void zero_kernel(unsigned int* __restrict__ p, size_t n) {
    size_t i = (size_t)blockIdx.x * blockDim.x + threadIdx.x;
    if (i < n) p[i] = 0u;
}

// ---------------- one fused LSTM step (one direction) --------------------------
// z = x_t @ Wi + h_in @ Wh + b ; gates ; masked state update.
// Block = 8 waves; wave -> (m0..m0+15) x (j0..j0+15) hidden patch, 4 gate accums.
// Weight tiles (shared by all 8 waves) are double-buffered through LDS with
// global_load_async_to_lds_b128; whole 64-K chunk is register-staged (2 A frags
// global + 8 B frags via ds_load_b128), then 8 WMMAs run back-to-back.
// Grid: x = H/16, y = B/128.
template<int KX>
__global__ __launch_bounds__(256)
void lstm_step_kernel(const unsigned short* __restrict__ xt,
                      const unsigned short* __restrict__ WiT,   // [4H, KX] bf16
                      const unsigned short* __restrict__ WhT,   // [4H, H]  bf16
                      const float* __restrict__ bias,           // [4H]
                      const unsigned short* __restrict__ h_in,  // [B, H] bf16
                      float* __restrict__ c_st,                 // [B, H] f32 (in-place)
                      unsigned short* __restrict__ h_out,       // [B, H] bf16
                      const unsigned char* __restrict__ mcol,   // [B]
                      unsigned short* __restrict__ ysrow,       // [B, ys_ld]+off or null
                      int ys_ld) {
    __shared__ __align__(16) unsigned short ldsB[2][4][16][KC];  // 16 KB

    const int lane = threadIdx.x;
    const int wave = threadIdx.y;
    const int tid  = wave * 32 + lane;
    const int m0   = (blockIdx.y * 8 + wave) * 16;
    const int j0   = blockIdx.x * 16;

    constexpr int NCI = KX / KC;           // chunks from input projection
    constexpr int NC  = NCI + HH / KC;     // + recurrent projection chunks

    // issue this thread's share of one chunk's async weight loads (8KB / block)
    auto stage = [&](int buf, const unsigned short* WT, int ldk, int kc) {
        #pragma unroll
        for (int r = 0; r < 2; ++r) {
            int c  = tid * 2 + r;              // 0..511 b128 chunks
            int g  = c >> 7;
            int n  = (c >> 3) & 15;
            int kk = (c & 7) * 8;
            unsigned loff = (unsigned)(uintptr_t)&ldsB[buf][g][n][kk];
            const unsigned short* gp = WT + (size_t)(g * HH + j0 + n) * ldk + kc + kk;
            async_ld_b128(loff, gp);
        }
    };
    auto chunk_src = [&](int ci, const unsigned short*& WT, int& ldk, int& kc,
                         const unsigned short*& Ab, int& lda) {
        if (ci < NCI) { WT = WiT; ldk = KX; kc = ci * KC;         Ab = xt;   lda = KX; }
        else          { WT = WhT; ldk = HH; kc = (ci - NCI) * KC; Ab = h_in; lda = HH; }
    };

    v8f acc[4];
    #pragma unroll
    for (int g = 0; g < 4; ++g) {
        float bv = bias[g * HH + j0 + (lane & 15)];
        #pragma unroll
        for (int v = 0; v < 8; ++v) acc[g][v] = bv;
    }

    // prologue: stage chunk 0
    {
        const unsigned short *WT, *Ab; int ldk, kc, lda;
        chunk_src(0, WT, ldk, kc, Ab, lda);
        stage(0, WT, ldk, kc);
    }
    wait_asynccnt0();
    __syncthreads();

    int buf = 0;
    const int kb16 = (lane >> 4) << 4;   // B-frag K offset within chunk half
    const int nsel = lane & 15;
    #pragma unroll 2
    for (int ci = 0; ci < NC; ++ci) {
        if (ci + 1 < NC) {
            const unsigned short *WT, *Ab; int ldk, kc, lda;
            chunk_src(ci + 1, WT, ldk, kc, Ab, lda);
            stage(buf ^ 1, WT, ldk, kc);
        }
        const unsigned short *WT, *Ab; int ldk, kc, lda;
        chunk_src(ci, WT, ldk, kc, Ab, lda);

        // register-stage the whole 64-wide chunk, then dense WMMA issue
        const unsigned short* abase = Ab + (size_t)m0 * lda + kc;
        __builtin_prefetch(abase + KC, 0, 1);
        v16bf a0 = load_a_frag(abase,      lda, lane);
        v16bf a1 = load_a_frag(abase + 32, lda, lane);
        v16bf bf[2][4];
        #pragma unroll
        for (int h = 0; h < 2; ++h) {
            #pragma unroll
            for (int g = 0; g < 4; ++g) {
                const uint4* q4 =
                    reinterpret_cast<const uint4*>(&ldsB[buf][g][nsel][h * 32 + kb16]);
                Q32 q; q.a = q4[0]; q.b = q4[1];
                bf[h][g] = __builtin_bit_cast(v16bf, q);
            }
        }
        #pragma unroll
        for (int g = 0; g < 4; ++g) acc[g] = wmma_bf16(a0, bf[0][g], acc[g]);
        #pragma unroll
        for (int g = 0; g < 4; ++g) acc[g] = wmma_bf16(a1, bf[1][g], acc[g]);

        wait_asynccnt0();
        __syncthreads();
        buf ^= 1;
    }

    const int jc  = j0 + nsel;
    const int hi8 = (lane >> 4) * 8;
    #pragma unroll
    for (int v = 0; v < 8; ++v) {
        int m = m0 + hi8 + v;
        float zi = acc[0][v], zf = acc[1][v], zg = acc[2][v], zo = acc[3][v];
        float cold = c_st[(size_t)m * HH + jc];
        float cn = sigf(zf) * cold + sigf(zi) * tanhf(zg);
        float hn = sigf(zo) * tanhf(cn);
        bool mk = (mcol[m] != 0);
        float hold = bf2f(h_in[(size_t)m * HH + jc]);
        float cw = mk ? cn : cold;
        float hw = mk ? hn : hold;
        c_st[(size_t)m * HH + jc]  = cw;
        unsigned short hb = f2bf(hw);
        h_out[(size_t)m * HH + jc] = hb;
        if (ysrow) ysrow[(size_t)m * ys_ld + jc] = hb;
    }
}

// ---------------- generic WMMA GEMM + bias + optional relu ---------------------
// Out[M,N](bf16) = act(A[M,K](bf16) @ BT[N,K]^T + bias). One wave per 16x16 tile.
__global__ __launch_bounds__(32)
void wmma_gemm_kernel(const unsigned short* __restrict__ A, int lda,
                      const unsigned short* __restrict__ BT,
                      const float* __restrict__ bias,
                      unsigned short* __restrict__ Out, int ldo,
                      int K, int relu) {
    const int lane = threadIdx.x;
    const int n0 = blockIdx.x * 16;
    const int m0 = blockIdx.y * 16;
    v8f acc;
    float bv = bias[n0 + (lane & 15)];
    #pragma unroll
    for (int v = 0; v < 8; ++v) acc[v] = bv;
    for (int k0 = 0; k0 < K; k0 += 32) {
        v16bf a = load_a_frag(A + (size_t)m0 * lda + k0, lda, lane);
        v16bf b = load_b_frag(BT, K, n0, k0, lane);
        acc = wmma_bf16(a, b, acc);
    }
    const int nc  = n0 + (lane & 15);
    const int hi8 = (lane >> 4) * 8;
    #pragma unroll
    for (int v = 0; v < 8; ++v) {
        int m = m0 + hi8 + v;
        float x = acc[v];
        if (relu) x = fmaxf(x, 0.0f);
        Out[(size_t)m * ldo + nc] = f2bf(x);
    }
}

// ---------------- features = [h2f | h2b | style_emb] ---------------------------
__global__ __launch_bounds__(256)
void feats_kernel(const unsigned short* __restrict__ h2f,
                  const unsigned short* __restrict__ h2b,
                  const int* __restrict__ styles,
                  const float* __restrict__ sty_emb,
                  unsigned short* __restrict__ feats) {
    int i = blockIdx.x * blockDim.x + threadIdx.x;
    if (i >= BB * FK) return;
    int b = i / FK, j = i % FK;
    unsigned short v;
    if (j < HH)            v = h2f[(size_t)b * HH + j];
    else if (j < 2 * HH)   v = h2b[(size_t)b * HH + (j - HH)];
    else                   v = f2bf(sty_emb[(size_t)styles[b] * STY + (j - 2 * HH)]);
    feats[i] = v;
}

// ---------------- final head: sigmoid(d2 @ W3 + b) -----------------------------
__global__ __launch_bounds__(256)
void head_kernel(const unsigned short* __restrict__ d2,
                 const float* __restrict__ W3, const float* __restrict__ bd3,
                 float* __restrict__ out) {
    int b = blockIdx.x * blockDim.x + threadIdx.x;
    if (b >= BB) return;
    float s = bd3[0];
    for (int k = 0; k < D2N; ++k) s += bf2f(d2[(size_t)b * D2N + k]) * W3[k];
    out[b] = 1.0f / (1.0f + __expf(-s));
}

extern "C" void kernel_launch(void* const* d_in, const int* in_sizes, int n_in,
                              void* d_out, int out_size, void* d_ws, size_t ws_size,
                              hipStream_t stream) {
    const int*   seq    = (const int*)  d_in[0];
    const int*   styles = (const int*)  d_in[1];
    const float* tok    = (const float*)d_in[2];
    const float* sty    = (const float*)d_in[3];
    const float* Wi1f = (const float*)d_in[4];
    const float* Wh1f = (const float*)d_in[5];
    const float* b1f  = (const float*)d_in[6];
    const float* Wi1b = (const float*)d_in[7];
    const float* Wh1b = (const float*)d_in[8];
    const float* b1b  = (const float*)d_in[9];
    const float* Wi2f = (const float*)d_in[10];
    const float* Wh2f = (const float*)d_in[11];
    const float* b2f  = (const float*)d_in[12];
    const float* Wi2b = (const float*)d_in[13];
    const float* Wh2b = (const float*)d_in[14];
    const float* b2b  = (const float*)d_in[15];
    const float* W1   = (const float*)d_in[16];
    const float* bd1  = (const float*)d_in[17];
    const float* W2   = (const float*)d_in[18];
    const float* bd2  = (const float*)d_in[19];
    const float* W3   = (const float*)d_in[20];
    const float* bd3  = (const float*)d_in[21];

    char* ws = (char*)d_ws;
    size_t off = 0;
    auto alloc = [&](size_t bytes) -> void* {
        void* p = ws + off;
        off = (off + bytes + 255) & ~(size_t)255;
        return p;
    };

    unsigned short* X1    = (unsigned short*)alloc((size_t)TT * BB * EMBD * 2);     // 64 MB
    unsigned short* ys1   = (unsigned short*)alloc((size_t)TT * BB * 1024 * 2);     // 128 MB
    unsigned char*  maskT = (unsigned char*) alloc((size_t)TT * BB);
    unsigned short* WiT1f = (unsigned short*)alloc((size_t)G4 * EMBD * 2);
    unsigned short* WhT1f = (unsigned short*)alloc((size_t)G4 * HH   * 2);
    unsigned short* WiT1b = (unsigned short*)alloc((size_t)G4 * EMBD * 2);
    unsigned short* WhT1b = (unsigned short*)alloc((size_t)G4 * HH   * 2);
    unsigned short* WiT2f = (unsigned short*)alloc((size_t)G4 * 1024 * 2);
    unsigned short* WhT2f = (unsigned short*)alloc((size_t)G4 * HH   * 2);
    unsigned short* WiT2b = (unsigned short*)alloc((size_t)G4 * 1024 * 2);
    unsigned short* WhT2b = (unsigned short*)alloc((size_t)G4 * HH   * 2);
    unsigned short* W1T   = (unsigned short*)alloc((size_t)D1N * FK  * 2);
    unsigned short* W2T   = (unsigned short*)alloc((size_t)D2N * D1N * 2);

    const size_t HSZ = (size_t)BB * HH;                 // state elements
    char* zbase = (char*)alloc(8 * HSZ * 2 + 4 * HSZ * 4);
    unsigned short* h1f_a = (unsigned short*)(zbase);
    unsigned short* h1f_b = h1f_a + HSZ;
    unsigned short* h1b_a = h1f_b + HSZ;
    unsigned short* h1b_b = h1b_a + HSZ;
    unsigned short* h2f_a = h1b_b + HSZ;
    unsigned short* h2f_b = h2f_a + HSZ;
    unsigned short* h2b_a = h2f_b + HSZ;
    unsigned short* h2b_b = h2b_a + HSZ;
    float* c1f = (float*)(h2b_b + HSZ);
    float* c1b = c1f + HSZ;
    float* c2f = c1b + HSZ;
    float* c2b = c2f + HSZ;

    unsigned short* feats = (unsigned short*)alloc((size_t)BB * FK  * 2);
    unsigned short* d1    = (unsigned short*)alloc((size_t)BB * D1N * 2);
    unsigned short* d2    = (unsigned short*)alloc((size_t)BB * D2N * 2);

    // --- 1. weight cast+transpose ---
    auto castT = [&](const float* in, unsigned short* out, int K, int N) {
        size_t n = (size_t)K * N;
        castT_kernel<<<dim3((unsigned)((n + 255) / 256)), dim3(256), 0, stream>>>(in, out, K, N);
    };
    castT(Wi1f, WiT1f, EMBD, G4);  castT(Wh1f, WhT1f, HH, G4);
    castT(Wi1b, WiT1b, EMBD, G4);  castT(Wh1b, WhT1b, HH, G4);
    castT(Wi2f, WiT2f, 1024, G4);  castT(Wh2f, WhT2f, HH, G4);
    castT(Wi2b, WiT2b, 1024, G4);  castT(Wh2b, WhT2b, HH, G4);
    castT(W1, W1T, FK, D1N);       castT(W2, W2T, D1N, D2N);

    // --- 2. embedding gather + mask ---
    {
        size_t n = (size_t)BB * TT * EMBD;
        embed_kernel<<<dim3((unsigned)((n + 255) / 256)), dim3(256), 0, stream>>>(seq, tok, X1, maskT);
    }

    // --- 3. zero states (h ping-pongs + c) ---
    {
        size_t nwords = (8 * HSZ * 2 + 4 * HSZ * 4) / 4;
        zero_kernel<<<dim3((unsigned)((nwords + 255) / 256)), dim3(256), 0, stream>>>((unsigned int*)zbase, nwords);
    }

    const dim3 lgrid(HH / 16, BB / 128);
    const dim3 lblk(32, 8);
    auto step1 = [&](const unsigned short* xt,
                     const unsigned short* WiT, const unsigned short* WhT, const float* bias,
                     const unsigned short* hin, float* c, unsigned short* hout,
                     const unsigned char* mc, unsigned short* ysr, int ysld) {
        lstm_step_kernel<EMBD><<<lgrid, lblk, 0, stream>>>(xt, WiT, WhT, bias, hin, c, hout, mc, ysr, ysld);
    };
    auto step2 = [&](const unsigned short* xt,
                     const unsigned short* WiT, const unsigned short* WhT, const float* bias,
                     const unsigned short* hin, float* c, unsigned short* hout,
                     const unsigned char* mc) {
        lstm_step_kernel<1024><<<lgrid, lblk, 0, stream>>>(xt, WiT, WhT, bias, hin, c, hout, mc, nullptr, 0);
    };

    // --- 4. layer 1 (fwd + bwd), ys1[t] = [h_f(t) | h_b(t)] ---
    for (int s = 0; s < TT; ++s) {
        int tf = s, tb = TT - 1 - s;
        const unsigned short* hfi = (s & 1) ? h1f_b : h1f_a;
        unsigned short*       hfo = (s & 1) ? h1f_a : h1f_b;
        const unsigned short* hbi = (s & 1) ? h1b_b : h1b_a;
        unsigned short*       hbo = (s & 1) ? h1b_a : h1b_b;
        step1(X1 + (size_t)tf * BB * EMBD, WiT1f, WhT1f, b1f, hfi, c1f, hfo,
              maskT + (size_t)tf * BB, ys1 + (size_t)tf * BB * 1024, 1024);
        step1(X1 + (size_t)tb * BB * EMBD, WiT1b, WhT1b, b1b, hbi, c1b, hbo,
              maskT + (size_t)tb * BB, ys1 + (size_t)tb * BB * 1024 + HH, 1024);
    }

    // --- 5. layer 2 (fwd + bwd), last state only ---
    for (int s = 0; s < TT; ++s) {
        int tf = s, tb = TT - 1 - s;
        const unsigned short* hfi = (s & 1) ? h2f_b : h2f_a;
        unsigned short*       hfo = (s & 1) ? h2f_a : h2f_b;
        const unsigned short* hbi = (s & 1) ? h2b_b : h2b_a;
        unsigned short*       hbo = (s & 1) ? h2b_a : h2b_b;
        step2(ys1 + (size_t)tf * BB * 1024, WiT2f, WhT2f, b2f, hfi, c2f, hfo,
              maskT + (size_t)tf * BB);
        step2(ys1 + (size_t)tb * BB * 1024, WiT2b, WhT2b, b2b, hbi, c2b, hbo,
              maskT + (size_t)tb * BB);
    }
    // after 256 steps (even number of swaps), final h lives in the _a buffers
    unsigned short* h2f_fin = h2f_a;
    unsigned short* h2b_fin = h2b_a;

    // --- 6. MLP head ---
    feats_kernel<<<dim3((BB * FK + 255) / 256), dim3(256), 0, stream>>>(h2f_fin, h2b_fin, styles, sty, feats);
    wmma_gemm_kernel<<<dim3(D1N / 16, BB / 16), dim3(32), 0, stream>>>(feats, FK, W1T, bd1, d1, D1N, FK, 1);
    wmma_gemm_kernel<<<dim3(D2N / 16, BB / 16), dim3(32), 0, stream>>>(d1, D1N, W2T, bd2, d2, D2N, D1N, 1);
    head_kernel<<<dim3(1), dim3(256), 0, stream>>>(d2, W3, bd3, (float*)d_out);

    (void)in_sizes; (void)n_in; (void)out_size; (void)ws_size;
}
